// EdgePredGNN_45346264711623
// MI455X (gfx1250) — matrix operand order repaired and checked by reading, not compiled, not run on hardware
//
#include <hip/hip_runtime.h>
#include <hip/hip_bf16.h>

typedef __attribute__((ext_vector_type(16))) _Float16 v16h;
typedef __attribute__((ext_vector_type(8)))  _Float16 v8h;
typedef __attribute__((ext_vector_type(8)))  float    v8f;

// ---------- helpers ----------
__device__ __forceinline__ unsigned fenc(float f) {
    unsigned u = __float_as_uint(f);
    return (u & 0x80000000u) ? ~u : (u | 0x80000000u);   // order-preserving encode
}
__device__ __forceinline__ float fdec(unsigned u) {
    return (u & 0x80000000u) ? __uint_as_float(u & 0x7FFFFFFFu) : __uint_as_float(~u);
}

// ---------- f32 -> f16 elementwise ----------
__global__ void cvt_f32_f16(const float* __restrict__ in, _Float16* __restrict__ out, long long n)
{
    long long i = (long long)blockIdx.x * blockDim.x + threadIdx.x;
    if (i < n) out[i] = (_Float16)in[i];
}

// ---------- f32 B[K][N] -> f16 Bt[N][Kpad], zero-padded in K ----------
__global__ void transpose_cvt(const float* __restrict__ B, _Float16* __restrict__ Bt,
                              int K, int N, int Kpad)
{
    int i = blockIdx.x * blockDim.x + threadIdx.x;
    if (i >= N * Kpad) return;
    int n = i / Kpad, k = i - n * Kpad;
    Bt[i] = (k < K) ? (_Float16)B[(size_t)k * N + n] : (_Float16)0.f;
}

// ---------- dense GEMM: C[M,N] = Ah[M,K] @ Bt[N,K]^T, f16 WMMA, f32 out ----------
// One wave computes a 16x64 strip (4 N-tiles share one A fragment).
// K % 32 == 0, N % 64 == 0. All fragment loads are contiguous 16B (v8h).
__global__ void wmma_gemm_f16(const _Float16* __restrict__ Ah, const _Float16* __restrict__ Bt,
                              float* __restrict__ C, int M, int N, int K)
{
    const int wavesPerBlock = blockDim.x >> 5;
    const int wave = blockIdx.x * wavesPerBlock + (threadIdx.x >> 5);
    const int lane = threadIdx.x & 31;
    const int groupsN = N >> 6;                // strips of 64 columns
    const int tm = wave / groupsN;
    const int tg = wave - tm * groupsN;
    if (tm * 16 >= M) return;                  // wave-uniform
    const int half = lane >> 4;                // 0: lanes 0-15, 1: lanes 16-31
    const int l    = lane & 15;
    int rowA = tm * 16 + l;
    if (rowA >= M) rowA = M - 1;               // clamp (harmless duplicate row)
    const _Float16* Ap = Ah + (size_t)rowA * K;

    v8f acc[4] = {};
    for (int k0 = 0; k0 < K; k0 += 32) {
        // A: lane's K-values live in two contiguous groups of 8 (ISA 7.12.2)
        v8h alo = *(const v8h*)(Ap + k0 + (half << 3));
        v8h ahi = *(const v8h*)(Ap + k0 + (half << 3) + 16);
        v16h a;
#pragma unroll
        for (int i = 0; i < 8; ++i) { a[i] = alo[i]; a[i + 8] = ahi[i]; }
#pragma unroll
        for (int j = 0; j < 4; ++j) {
            int col = (tg * 4 + j) * 16 + l;
            const _Float16* Bp = Bt + (size_t)col * K + k0 + (half << 3);
            v8h blo = *(const v8h*)(Bp);
            v8h bhi = *(const v8h*)(Bp + 16);
            v16h b;
#pragma unroll
            for (int i = 0; i < 8; ++i) { b[i] = blo[i]; b[i + 8] = bhi[i]; }
            acc[j] = __builtin_amdgcn_wmma_f32_16x16x32_f16(
                false, a, false, b, (short)0, acc[j], false, false);
        }
    }
#pragma unroll
    for (int j = 0; j < 4; ++j) {
        int col = (tg * 4 + j) * 16 + l;
#pragma unroll
        for (int r = 0; r < 8; ++r) {
            int row = tm * 16 + r + half * 8;
            if (row < M) C[(size_t)row * N + col] = acc[j][r];
        }
    }
}

// ---------- attention logits: als/ald[n,h] = sum_c lin[n,h,c] * a{s,d}[h,c] ----------
__global__ void attn_logits(const float* __restrict__ lin,
                            const float* __restrict__ as_, const float* __restrict__ ad_,
                            float* __restrict__ als, float* __restrict__ ald,
                            int Nn, int H, int C)
{
    int t = blockIdx.x * blockDim.x + threadIdx.x;
    if (t >= Nn * H) return;
    int n = t / H, h = t - n * H;
    const float4* hp = (const float4*)(lin + (size_t)n * H * C + (size_t)h * C);
    const float4* sp = (const float4*)(as_ + h * C);
    const float4* dp = (const float4*)(ad_ + h * C);
    float s = 0.f, d = 0.f;
    for (int c = 0; c < (C >> 2); ++c) {
        float4 v = hp[c], a = sp[c], b = dp[c];
        s += v.x * a.x + v.y * a.y + v.z * a.z + v.w * a.w;
        d += v.x * b.x + v.y * b.y + v.z * b.z + v.w * b.w;
    }
    als[t] = s;
    ald[t] = d;
}

// ---------- edge pass 1: segment max of leaky_relu(als[src]+ald[dst]) ----------
__global__ void edge_pass_max(const int* __restrict__ ei, int E, int Nn, int H,
                              const float* __restrict__ als, const float* __restrict__ ald,
                              unsigned* __restrict__ mmax)
{
    int t = blockIdx.x * blockDim.x + threadIdx.x;
    int total = (E + Nn) * H;
    if (t >= total) return;
    int e = t / H, h = t - e * H;
    int src = (e < E) ? ei[e]     : (e - E);   // self loops appended
    int dst = (e < E) ? ei[E + e] : (e - E);
    float v = als[src * H + h] + ald[dst * H + h];
    v = (v > 0.f) ? v : 0.2f * v;              // leaky_relu, slope 0.2
    atomicMax(&mmax[dst * H + h], fenc(v));
}

// ---------- edge pass 2: z[dst,h] += exp(e - m[dst,h]) ----------
__global__ void edge_pass_sum(const int* __restrict__ ei, int E, int Nn, int H,
                              const float* __restrict__ als, const float* __restrict__ ald,
                              const unsigned* __restrict__ mmax, float* __restrict__ zsum)
{
    int t = blockIdx.x * blockDim.x + threadIdx.x;
    int total = (E + Nn) * H;
    if (t >= total) return;
    int e = t / H, h = t - e * H;
    int src = (e < E) ? ei[e]     : (e - E);
    int dst = (e < E) ? ei[E + e] : (e - E);
    int idx = dst * H + h;
    float v = als[src * H + h] + ald[idx];
    v = (v > 0.f) ? v : 0.2f * v;
    atomicAdd(&zsum[idx], expf(v - fdec(mmax[idx])));
}

// ---------- edge pass 3: out[dst,h,c] += alpha * lin[src,h,c]; one block per edge ----------
// blockDim == H*C/4; each thread handles 4 channels (float4 gather + 4 atomics)
__global__ void edge_aggregate(const int* __restrict__ ei, int E, int Nn, int H, int C,
                               const float* __restrict__ lin,
                               const float* __restrict__ als, const float* __restrict__ ald,
                               const unsigned* __restrict__ mmax, const float* __restrict__ zsum,
                               float* __restrict__ out)
{
    __shared__ float alpha_s[4];
    int e = blockIdx.x;
    int tid = threadIdx.x;
    int C4 = C >> 2;
    int h = tid / C4, c4 = tid - h * C4;
    int src = (e < E) ? ei[e]     : (e - E);
    int dst = (e < E) ? ei[E + e] : (e - E);
    if (c4 == 0) {
        int idx = dst * H + h;
        float v = als[src * H + h] + ald[idx];
        v = (v > 0.f) ? v : 0.2f * v;
        float w = expf(v - fdec(mmax[idx]));
        alpha_s[h] = w / zsum[idx];
    }
    __syncthreads();
    const float4* lp = (const float4*)(lin + (size_t)src * H * C);
    float4 v = lp[tid];
    float a = alpha_s[h];
    float* op = out + (size_t)dst * H * C + (size_t)tid * 4;
    atomicAdd(op + 0, v.x * a);
    atomicAdd(op + 1, v.y * a);
    atomicAdd(op + 2, v.z * a);
    atomicAdd(op + 3, v.w * a);
}

// ---------- bias (+ optional ELU) epilogue ----------
__global__ void bias_act(float* __restrict__ y, const float* __restrict__ b,
                         long long total, int D, int act)
{
    long long i = (long long)blockIdx.x * blockDim.x + threadIdx.x;
    if (i >= total) return;
    int d = (int)(i % D);
    float v = y[i] + b[d];
    if (act == 1) v = (v > 0.f) ? v : (expf(v) - 1.f);  // ELU
    y[i] = v;
}

// ---------- fused edge MLP: feat[E,258] -> 64 -> 32 -> 1, WMMA in LDS ----------
#define MLP_K 288   // 258 padded up to multiple of 32
__global__ __launch_bounds__(256) void edge_mlp(
    const float* __restrict__ h3, const int* __restrict__ ei, const float* __restrict__ ea,
    const _Float16* __restrict__ mw1t,   // [64][288] f16, zero-padded transpose of mw1
    const float* __restrict__ mb1,
    const _Float16* __restrict__ mw2t,   // [32][64] f16 transpose of mw2
    const float* __restrict__ mb2,
    const float* __restrict__ mw3, const float* __restrict__ mb3,
    float* __restrict__ out, int E)
{
    __shared__ _Float16 featT[64][MLP_K];   // 36864 B
    __shared__ _Float16 o1h[64][64];        //  8192 B
    __shared__ float    o2f[64][32];        //  8192 B
    const int e0 = blockIdx.x * 64;
    const int tid = threadIdx.x;

    // stage feat = [h3[row] | h3[col] | edge_attr | 0-pad] as f16
    for (int i = tid; i < 64 * MLP_K; i += 256) {
        int r = i / MLP_K, k = i - r * MLP_K;
        int e = e0 + r;
        float v = 0.f;
        if (e < E) {
            if (k < 128)       v = h3[(size_t)ei[e] * 128 + k];
            else if (k < 256)  v = h3[(size_t)ei[E + e] * 128 + (k - 128)];
            else if (k < 258)  v = ea[(size_t)e * 2 + (k - 256)];
        }
        featT[r][k] = (_Float16)v;
    }
    __syncthreads();

    const int wave = tid >> 5, lane = tid & 31;
    const int half = lane >> 4, l = lane & 15;

    // o1 = relu(feat @ mw1 + mb1): M=64,N=64,K=288 -> 16 tiles over 8 waves
    for (int t = wave; t < 16; t += 8) {
        int tm = t >> 2, tn = t & 3;
        int rowA = tm * 16 + l;
        int colB = tn * 16 + l;
        v8f c = {};
        for (int k0 = 0; k0 < MLP_K; k0 += 32) {
            v8h alo = *(const v8h*)(&featT[rowA][k0 + (half << 3)]);
            v8h ahi = *(const v8h*)(&featT[rowA][k0 + (half << 3) + 16]);
            const _Float16* Bp = mw1t + (size_t)colB * MLP_K + k0 + (half << 3);
            v8h blo = *(const v8h*)(Bp);
            v8h bhi = *(const v8h*)(Bp + 16);
            v16h a, b;
#pragma unroll
            for (int i = 0; i < 8; ++i) {
                a[i] = alo[i]; a[i + 8] = ahi[i];
                b[i] = blo[i]; b[i + 8] = bhi[i];
            }
            c = __builtin_amdgcn_wmma_f32_16x16x32_f16(false, a, false, b, (short)0, c, false, false);
        }
        float bias = mb1[colB];
#pragma unroll
        for (int r = 0; r < 8; ++r) {
            int row = tm * 16 + r + half * 8;
            o1h[row][colB] = (_Float16)fmaxf(c[r] + bias, 0.f);
        }
    }
    __syncthreads();

    // o2 = relu(o1 @ mw2 + mb2): M=64,N=32,K=64 -> 8 tiles, one per wave
    {
        int tm = wave >> 1, tn = wave & 1;
        int rowA = tm * 16 + l;
        int colB = tn * 16 + l;
        v8f c = {};
        for (int k0 = 0; k0 < 64; k0 += 32) {
            v8h alo = *(const v8h*)(&o1h[rowA][k0 + (half << 3)]);
            v8h ahi = *(const v8h*)(&o1h[rowA][k0 + (half << 3) + 16]);
            const _Float16* Bp = mw2t + (size_t)colB * 64 + k0 + (half << 3);
            v8h blo = *(const v8h*)(Bp);
            v8h bhi = *(const v8h*)(Bp + 16);
            v16h a, b;
#pragma unroll
            for (int i = 0; i < 8; ++i) {
                a[i] = alo[i]; a[i + 8] = ahi[i];
                b[i] = blo[i]; b[i + 8] = bhi[i];
            }
            c = __builtin_amdgcn_wmma_f32_16x16x32_f16(false, a, false, b, (short)0, c, false, false);
        }
        float bias = mb2[colB];
#pragma unroll
        for (int r = 0; r < 8; ++r) {
            int row = tm * 16 + r + half * 8;
            o2f[row][colB] = fmaxf(c[r] + bias, 0.f);
        }
    }
    __syncthreads();

    // o3 = o2 @ mw3 + mb3 (scalar dot per edge)
    if (tid < 64) {
        int e = e0 + tid;
        if (e < E) {
            float acc = mb3[0];
#pragma unroll
            for (int k = 0; k < 32; ++k) acc += o2f[tid][k] * mw3[k];
            out[e] = acc;
        }
    }
}

// ---------- host driver ----------
extern "C" void kernel_launch(void* const* d_in, const int* in_sizes, int n_in,
                              void* d_out, int out_size, void* d_ws, size_t ws_size,
                              hipStream_t stream) {
    const float* x   = (const float*)d_in[0];
    const int*   ei  = (const int*)d_in[1];
    const float* ea  = (const float*)d_in[2];
    const float* W1  = (const float*)d_in[3];
    const float* a1s = (const float*)d_in[4];
    const float* a1d = (const float*)d_in[5];
    const float* b1  = (const float*)d_in[6];
    const float* W2  = (const float*)d_in[7];
    const float* a2s = (const float*)d_in[8];
    const float* a2d = (const float*)d_in[9];
    const float* b2  = (const float*)d_in[10];
    const float* W3  = (const float*)d_in[11];
    const float* a3s = (const float*)d_in[12];
    const float* a3d = (const float*)d_in[13];
    const float* b3  = (const float*)d_in[14];
    const float* mw1 = (const float*)d_in[15];
    const float* mb1 = (const float*)d_in[16];
    const float* mw2 = (const float*)d_in[17];
    const float* mb2 = (const float*)d_in[18];
    const float* mw3 = (const float*)d_in[19];
    const float* mb3 = (const float*)d_in[20];
    (void)n_in; (void)out_size; (void)ws_size;

    const int Nn = in_sizes[0] / 32;   // 50000
    const int E  = in_sizes[1] / 2;    // 800000

    // workspace carve-out
    char* ws = (char*)d_ws;
    size_t off = 0;
    auto alloc = [&](size_t bytes) -> void* {
        void* p = (void*)(ws + off);
        off += (bytes + 255) & ~(size_t)255;
        return p;
    };
    float*     lin  = (float*)alloc((size_t)Nn * 256 * 4);
    float*     h1   = (float*)alloc((size_t)Nn * 256 * 4);
    float*     h2   = (float*)alloc((size_t)Nn * 256 * 4);
    float*     h3   = (float*)alloc((size_t)Nn * 128 * 4);
    _Float16*  Ah   = (_Float16*)alloc((size_t)Nn * 256 * 2);  // f16 activations
    float*     als  = (float*)alloc((size_t)Nn * 4 * 4);
    float*     ald  = (float*)alloc((size_t)Nn * 4 * 4);
    unsigned*  mmax = (unsigned*)alloc((size_t)Nn * 4 * 4);
    float*     zsum = (float*)alloc((size_t)Nn * 4 * 4);
    _Float16*  W1t  = (_Float16*)alloc((size_t)256 * 32 * 2);
    _Float16*  W2t  = (_Float16*)alloc((size_t)256 * 256 * 2);
    _Float16*  W3t  = (_Float16*)alloc((size_t)128 * 256 * 2);
    _Float16*  mw1t = (_Float16*)alloc((size_t)64 * MLP_K * 2);
    _Float16*  mw2t = (_Float16*)alloc((size_t)32 * 64 * 2);

    // pre-convert weights to f16 transposed [N][Kpad]
    auto tc = [&](const float* B, _Float16* Bt, int K, int N, int Kpad) {
        int n = N * Kpad;
        transpose_cvt<<<(n + 255) / 256, 256, 0, stream>>>(B, Bt, K, N, Kpad);
    };
    tc(W1,  W1t,  32,  256, 32);
    tc(W2,  W2t,  256, 256, 256);
    tc(W3,  W3t,  256, 128, 256);
    tc(mw1, mw1t, 258, 64,  MLP_K);
    tc(mw2, mw2t, 64,  32,  64);

    const int E2 = E + Nn;

    auto run_gat = [&](const float* in, int K, const _Float16* Wt,
                       const float* as_, const float* ad_, const float* bb,
                       int H, int C, float* outb, int act) {
        const int D = H * C;
        {   // Ah = f16(in); lin = Ah @ Wt^T  (WMMA)
            long long n = (long long)Nn * K;
            cvt_f32_f16<<<(int)((n + 255) / 256), 256, 0, stream>>>(in, Ah, n);
            int jobs = ((Nn + 15) / 16) * (D / 64);
            int blocks = (jobs + 3) / 4;
            wmma_gemm_f16<<<blocks, 128, 0, stream>>>(Ah, Wt, lin, Nn, D, K);
        }
        attn_logits<<<(Nn * H + 255) / 256, 256, 0, stream>>>(lin, as_, ad_, als, ald, Nn, H, C);
        hipMemsetAsync(mmax, 0, (size_t)Nn * H * 4, stream);  // 0 encodes < -inf
        hipMemsetAsync(zsum, 0, (size_t)Nn * H * 4, stream);
        hipMemsetAsync(outb, 0, (size_t)Nn * D * 4, stream);
        int tot = E2 * H;
        edge_pass_max<<<(tot + 255) / 256, 256, 0, stream>>>(ei, E, Nn, H, als, ald, mmax);
        edge_pass_sum<<<(tot + 255) / 256, 256, 0, stream>>>(ei, E, Nn, H, als, ald, mmax, zsum);
        edge_aggregate<<<E2, D / 4, 0, stream>>>(ei, E, Nn, H, C, lin, als, ald, mmax, zsum, outb);
        long long total = (long long)Nn * D;
        bias_act<<<(int)((total + 255) / 256), 256, 0, stream>>>(outb, bb, total, D, act);
    };

    run_gat(x,  32,  W1t, a1s, a1d, b1, 4, 64,  h1, 1);   // conv1 + ELU
    run_gat(h1, 256, W2t, a2s, a2d, b2, 4, 64,  h2, 1);   // conv2 + ELU
    run_gat(h2, 256, W3t, a3s, a3d, b3, 1, 128, h3, 0);   // conv3 (H=1, mean==identity)

    edge_mlp<<<(E + 63) / 64, 256, 0, stream>>>(h3, ei, ea, mw1t, mb1, mw2t, mb2, mw3, mb3,
                                                (float*)d_out, E);
}
